// BinarizedDense_18889266168358
// MI455X (gfx1250) — matrix operand order repaired and checked by reading, not compiled, not run on hardware
//
#include <hip/hip_runtime.h>

typedef __attribute__((ext_vector_type(16))) __bf16 v16bf;
typedef __attribute__((ext_vector_type(8)))  __bf16 bf16x8;
typedef __attribute__((ext_vector_type(4)))  __bf16 bf16x4;
typedef __attribute__((ext_vector_type(8)))  float  v8f;
typedef __attribute__((ext_vector_type(4)))  float  f32x4;

#define TOKENS 8192
#define IN_F   4096
#define OUT_F  4096

#define BM 128
#define BN 128
#define BK 32
#define KTILES (IN_F / BK)

union AFrag { v16bf v; bf16x8 h[2]; };

__global__ __launch_bounds__(256)
void binarized_dense_wmma(const float* __restrict__ A,
                          const float* __restrict__ W,
                          float* __restrict__ C) {
    // Double-buffered LDS: A tile as bf16 hi/lo planes, W tile binarized+transposed [n][k]
    __shared__ __align__(16) __bf16 sAhi[2][BM * BK];   // 16 KB
    __shared__ __align__(16) __bf16 sAlo[2][BM * BK];   // 16 KB
    __shared__ __align__(16) __bf16 sBt [2][BN * BK];   // 16 KB

    const int t      = threadIdx.x;
    const int lane   = t & 31;
    const int wave   = t >> 5;
    const int wm     = wave >> 1;            // 0..3 -> M offset wm*32
    const int wn     = wave & 1;             // 0..1 -> N offset wn*64
    const size_t blockM = (size_t)blockIdx.y * BM;
    const size_t blockN = (size_t)blockIdx.x * BN;

    const int laneHalf = lane >> 4;
    const int l15      = lane & 15;

    // Per-thread 4(K)x4(N) patch of the W tile (register transpose, coalesced loads)
    const int pk = (t >> 5) * 4;             // k row 0,4,...,28
    const int pn = (t & 31) * 4;             // n col 0,4,...,124

    v8f acc[2][4] = {};

    f32x4 aReg[4];
    f32x4 bReg[4];

    auto load_tile = [&](int kt) {
        const int k0 = kt * BK;
        #pragma unroll
        for (int i = 0; i < 4; ++i) {
            const int li  = (i * 256 + t) * 4;      // 0..4095
            const int row = li >> 5;                // 0..127
            const int col = li & 31;
            aReg[i] = *(const f32x4*)(A + (blockM + row) * IN_F + k0 + col);
        }
        #pragma unroll
        for (int r = 0; r < 4; ++r)
            bReg[r] = *(const f32x4*)(W + (size_t)(k0 + pk + r) * OUT_F + blockN + pn);
    };

    auto store_tile = [&](int buf) {
        #pragma unroll
        for (int i = 0; i < 4; ++i) {
            const int li  = (i * 256 + t) * 4;
            const int row = li >> 5;
            const int col = li & 31;
            bf16x4 hi4, lo4;
            #pragma unroll
            for (int j = 0; j < 4; ++j) {
                float  x = aReg[i][j];
                __bf16 h = (__bf16)x;
                hi4[j] = h;
                lo4[j] = (__bf16)(x - (float)h);
            }
            *(bf16x4*)(&sAhi[buf][row * BK + col]) = hi4;
            *(bf16x4*)(&sAlo[buf][row * BK + col]) = lo4;
        }
        // binarize + register transpose: 4 x ds_store_b64 instead of 16 x ds_store_b16
        #pragma unroll
        for (int j = 0; j < 4; ++j) {
            bf16x4 v;
            #pragma unroll
            for (int r = 0; r < 4; ++r)
                v[r] = (bReg[r][j] > 0.5f) ? (__bf16)1.0f : (__bf16)0.0f;
            *(bf16x4*)(&sBt[buf][(pn + j) * BK + pk]) = v;
        }
    };

    load_tile(0);
    store_tile(0);
    __syncthreads();

    for (int kt = 0; kt < KTILES; ++kt) {
        const int buf = kt & 1;

        if (kt + 1 < KTILES) load_tile(kt + 1);    // overlap global loads with WMMA
        if (kt + 2 < KTILES) {
            __builtin_prefetch(A + (blockM + (t >> 1)) * IN_F + (kt + 2) * BK + (t & 1) * 16, 0, 3);
            __builtin_prefetch(W + (size_t)((kt + 2) * BK + pk) * OUT_F + blockN + pn, 0, 3);
        }

        // A fragments (both M sub-tiles, hi+lo planes)
        AFrag ahi[2], alo[2];
        #pragma unroll
        for (int tm = 0; tm < 2; ++tm) {
            const int mRow = wm * 32 + tm * 16 + l15;
            const int kb   = laneHalf * 8;
            ahi[tm].h[0] = *(const bf16x8*)(&sAhi[buf][mRow * BK + kb]);
            ahi[tm].h[1] = *(const bf16x8*)(&sAhi[buf][mRow * BK + kb + 16]);
            alo[tm].h[0] = *(const bf16x8*)(&sAlo[buf][mRow * BK + kb]);
            alo[tm].h[1] = *(const bf16x8*)(&sAlo[buf][mRow * BK + kb + 16]);
        }

        #pragma unroll
        for (int tn = 0; tn < 4; ++tn) {
            const int nCol = wn * 64 + tn * 16 + l15;
            AFrag bfr;
            bfr.h[0] = *(const bf16x8*)(&sBt[buf][nCol * BK + laneHalf * 8]);
            bfr.h[1] = *(const bf16x8*)(&sBt[buf][nCol * BK + laneHalf * 8 + 16]);
            #pragma unroll
            for (int tm = 0; tm < 2; ++tm) {
                acc[tm][tn] = __builtin_amdgcn_wmma_f32_16x16x32_bf16(
                    false, ahi[tm].v, false, bfr.v, (short)0, acc[tm][tn], false, false);
                acc[tm][tn] = __builtin_amdgcn_wmma_f32_16x16x32_bf16(
                    false, alo[tm].v, false, bfr.v, (short)0, acc[tm][tn], false, false);
            }
        }

        if (kt + 1 < KTILES) store_tile((kt + 1) & 1);
        __syncthreads();
    }

    // Store C: VGPR r -> M = r (lanes 0-15) / r+8 (lanes 16-31), N = lane&15
    #pragma unroll
    for (int tm = 0; tm < 2; ++tm) {
        #pragma unroll
        for (int tn = 0; tn < 4; ++tn) {
            const size_t mBase = blockM + wm * 32 + tm * 16 + laneHalf * 8;
            const size_t nIdx  = blockN + wn * 64 + tn * 16 + l15;
            #pragma unroll
            for (int r = 0; r < 8; ++r)
                C[(mBase + r) * OUT_F + nIdx] = acc[tm][tn][r];
        }
    }
}

extern "C" void kernel_launch(void* const* d_in, const int* in_sizes, int n_in,
                              void* d_out, int out_size, void* d_ws, size_t ws_size,
                              hipStream_t stream) {
    (void)in_sizes; (void)n_in; (void)out_size; (void)d_ws; (void)ws_size;
    const float* A = (const float*)d_in[0];   // [TOKENS, IN_F]  f32
    const float* W = (const float*)d_in[1];   // [IN_F, OUT_F]   f32 uniform[0,1)
    float*       C = (float*)d_out;           // [TOKENS, OUT_F] f32

    dim3 grid(OUT_F / BN, TOKENS / BM);       // (32, 64)
    dim3 block(256);
    binarized_dense_wmma<<<grid, block, 0, stream>>>(A, W, C);
}
